// dcell_nn_24747601559944
// MI455X (gfx1250) — matrix operand order repaired and checked by reading, not compiled, not run on hardware
//
#include <hip/hip_runtime.h>
#include <stdint.h>

// ---------------------------------------------------------------------------
// Trivial ext-vector types (union-safe, load/store as b128)
// ---------------------------------------------------------------------------
typedef __bf16    v16bf __attribute__((ext_vector_type(16)));
typedef float     v8f   __attribute__((ext_vector_type(8)));
typedef float     v4f   __attribute__((ext_vector_type(4)));
typedef uint32_t  v4u   __attribute__((ext_vector_type(4)));

#define BN_EPS 1e-5f

// Model dims
#define BATCH 256
#define NGENE 3008
#define NTOT  20480      // 2048 terms * 10 genes (flattened N of the big GEMM)
#define OUTW  2338       // 2048+256+32+1+1 output columns

// ---------------------------------------------------------------------------
// CDNA5 async global->LDS path (ASYNCcnt), guarded so the file builds either
// way. Probe round 2 confirmed the builtin exists with signature
//   (int4 AS1* src, int4 AS3* dst, imm int offset, imm int cpol)
// ---------------------------------------------------------------------------
#if defined(__has_builtin)
# if __has_builtin(__builtin_amdgcn_global_load_async_to_lds_b128)
#  define HAVE_ASYNC_LDS 1
# endif
#endif
#ifndef HAVE_ASYNC_LDS
# define HAVE_ASYNC_LDS 0
#endif

#if HAVE_ASYNC_LDS
typedef int v4i_ __attribute__((vector_size(4 * sizeof(int))));
typedef __attribute__((address_space(1))) v4i_ as1_v4i;
typedef __attribute__((address_space(3))) v4i_ as3_v4i;
__device__ __forceinline__ void async_copy16(const void* g, void* l) {
    // generic->AS1: full 64-bit VA; generic->AS3: low 32 bits are the LDS offset
    __builtin_amdgcn_global_load_async_to_lds_b128(
        (as1_v4i*)(uintptr_t)g, (as3_v4i*)(uint32_t)(uintptr_t)l, 0, 0);
}
__device__ __forceinline__ void wait_async0() {
# if __has_builtin(__builtin_amdgcn_s_wait_asynccnt)
    __builtin_amdgcn_s_wait_asynccnt(0);
# else
    asm volatile("s_wait_asynccnt 0" ::: "memory");
# endif
}
#endif

// one-instruction fp32x2 -> bf16x2 pack (v_perm_b32, truncation)
__device__ __forceinline__ uint32_t pk2(float lo, float hi) {
    return __builtin_amdgcn_perm(__float_as_uint(hi), __float_as_uint(lo), 0x07060302u);
}

// fp32 -> bf16 with round-to-nearest-even (used only in the one-shot x convert)
__device__ __forceinline__ uint32_t pk2_rne(float lo, float hi) {
    uint32_t ulo = __float_as_uint(lo);
    uint32_t uhi = __float_as_uint(hi);
    ulo = (ulo + 0x7FFFu + ((ulo >> 16) & 1u)) >> 16;
    uhi = (uhi + 0x7FFFu + ((uhi >> 16) & 1u)) & 0xFFFF0000u;
    return (ulo & 0xFFFFu) | uhi;
}

// ---------------------------------------------------------------------------
// Kernel 0: one-shot x [256 x 3008] fp32 -> bf16 (1.5 MB), so the GEMM's
// A-operand staging is pure data movement. Grid 376 x 256, 8 floats/thread.
// ---------------------------------------------------------------------------
__global__ __launch_bounds__(256) void xcvt_kernel(const float* __restrict__ x,
                                                   uint32_t* __restrict__ xbf)
{
    const size_t i = ((size_t)blockIdx.x * 256 + threadIdx.x) * 8;
    const v4f* p = (const v4f*)(x + i);
    v4f f0 = p[0], f1 = p[1];
    v4u o;
    o[0] = pk2_rne(f0[0], f0[1]); o[1] = pk2_rne(f0[2], f0[3]);
    o[2] = pk2_rne(f1[0], f1[1]); o[3] = pk2_rne(f1[2], f1[3]);
    *(v4u*)(xbf + i / 2) = o;
}

// ---------------------------------------------------------------------------
// Kernel 1: leaf gene layer as one big GEMM
//   h0[b, n] = sum_k x[b,k] * gW[n,k] + gb[n]
// A = xbf [256 x 3008] bf16 (async-copied to LDS), B = gW [20480 x 3008]
// fp32 streamed once from HBM and packed to bf16 with v_perm_b32.
// Grid: 160 blocks x 256 threads; block = 128 N cols x all 256 M rows;
// K chunk = 64 (two 16x16x32 WMMAs per M-tile per chunk).
// ---------------------------------------------------------------------------
__global__ __launch_bounds__(256) void gene_gemm_kernel(
    const uint32_t* __restrict__ xbf, const float* __restrict__ gW,
    const float* __restrict__ gb, float* __restrict__ h0)
{
    constexpr int NBLK = 128;   // N columns per workgroup
    constexpr int KCH  = 64;    // K per iteration
    constexpr int ROWP = 72;    // halves per LDS row (64 + 8 pad, 16B-aligned rows)

    __shared__ __align__(16) __bf16 sA[BATCH * ROWP];  // 36864 B
    __shared__ __align__(16) __bf16 sB[NBLK * ROWP];   // 18432 B

    const int tid  = threadIdx.x;
    const int lane = tid & 31;
    const int wv   = tid >> 5;      // wave 0..7 -> owns 16 N columns
    const int ln   = lane & 15;
    const int hf   = lane >> 4;     // half-wave select
    const int nblock0 = blockIdx.x * NBLK;

    v8f acc[16];
    const v8f vzero = {0.f,0.f,0.f,0.f,0.f,0.f,0.f,0.f};
#pragma unroll
    for (int i = 0; i < 16; ++i) acc[i] = vzero;

    // staging assignments: A -> one bf16 row (128B/chunk) per thread,
    //                      B -> half of one gW row (32 floats/chunk) per thread
    const int brow  = tid >> 1;           // 0..127
    const int bhalf = (tid & 1) * 32;     // float (== half) offset inside chunk
    const char*  xrow = (const char*)(xbf + (size_t)tid * (NGENE / 2));
    const float* grow = gW + (size_t)(nblock0 + brow) * NGENE + bhalf;
    char* sArow = (char*)(sA + tid * ROWP);
    v4u*  sBq   = (v4u*)(sB + brow * ROWP + bhalf);

    for (int k = 0; k < NGENE; k += KCH) {
        // ---- prefetch B chunk (fp32) into registers
        v4f bf_[8];
        const v4f* b4 = (const v4f*)(grow + k);
#pragma unroll
        for (int j = 0; j < 8; ++j) bf_[j] = b4[j];
#if !HAVE_ASYNC_LDS
        v4u af_[8];
        const v4u* a4 = (const v4u*)(xrow + 2 * k);
#pragma unroll
        for (int j = 0; j < 8; ++j) af_[j] = a4[j];
#endif
        __syncthreads();   // previous iteration's LDS reads complete

#if HAVE_ASYNC_LDS
        // A chunk: 128B per thread, DMA'd straight into LDS (ASYNCcnt)
#pragma unroll
        for (int j = 0; j < 8; ++j)
            async_copy16(xrow + 2 * k + 16 * j, sArow + 16 * j);
#else
#pragma unroll
        for (int j = 0; j < 8; ++j) ((v4u*)sArow)[j] = af_[j];
#endif
        // B chunk: pack fp32 -> bf16x2 with one v_perm_b32 each
#pragma unroll
        for (int j = 0; j < 4; ++j) {
            v4u t;
            t[0] = pk2(bf_[2*j][0],   bf_[2*j][1]);
            t[1] = pk2(bf_[2*j][2],   bf_[2*j][3]);
            t[2] = pk2(bf_[2*j+1][0], bf_[2*j+1][1]);
            t[3] = pk2(bf_[2*j+1][2], bf_[2*j+1][3]);
            sBq[j] = t;
        }
#if HAVE_ASYNC_LDS
        wait_async0();     // drain this wave's async copies before signaling
#endif
        __syncthreads();

        // ---- two 16x16x32 WMMA sweeps over the 64-wide K chunk
        union { v4u q[2]; v16bf v; } fbf, faf;
        const int koff = hf * 16;
#pragma unroll
        for (int kh = 0; kh < 2; ++kh) {
            // B frag: lane = column (wv*16+ln), K halves [kh*32 + hf*16, +16)
            const __bf16* bbase = sB + (wv*16 + ln) * ROWP + kh*32 + hf*16;
            fbf.q[0] = *(const v4u*)(bbase);
            fbf.q[1] = *(const v4u*)(bbase + 8);
#pragma unroll
            for (int mt = 0; mt < 16; ++mt) {
                // A frag: lane = row, K halves [kh*32 + hf*16, +16)
                const __bf16* abase = sA + (mt*16 + ln) * ROWP + kh*32 + koff;
                faf.q[0] = *(const v4u*)(abase);
                faf.q[1] = *(const v4u*)(abase + 8);
                acc[mt] = __builtin_amdgcn_wmma_f32_16x16x32_bf16(
                    false, faf.v, false, fbf.v, (short)0, acc[mt], false, false);
            }
        }
    }

    // ---- epilogue: D layout lane->N, vgpr i -> M = mt*16 + hf*8 + i
    const int n = nblock0 + wv*16 + ln;
    const float bias = gb[n];
#pragma unroll
    for (int mt = 0; mt < 16; ++mt) {
        const int m0 = mt*16 + hf*8;
#pragma unroll
        for (int i = 0; i < 8; ++i)
            h0[(size_t)(m0 + i) * NTOT + n] = acc[mt][i] + bias;
    }
}

// ---------------------------------------------------------------------------
// Kernel 2: one DAG level, fused linear + tanh + BatchNorm(train) + aux head.
// One block per term, one thread per batch row (256 == BATCH). BN stats via
// wave32 shfl-xor tree + ds_add_f32 LDS atomics.
// ---------------------------------------------------------------------------
template<int IN>
__global__ __launch_bounds__(256) void level_kernel(
    const float* __restrict__ hin,   // [256, T*IN]
    const float* __restrict__ w,     // [T, 6, IN]
    const float* __restrict__ bias,  // [T, 6]
    const float* __restrict__ gamma, const float* __restrict__ beta,
    const float* __restrict__ p, const float* __restrict__ q,
    const float* __restrict__ r, const float* __restrict__ s,
    float* __restrict__ hout,        // [256, T*6]
    float* __restrict__ out,         // [256, OUTW]
    int T, int auxcol)
{
    constexpr int H = 6;
    __shared__ float sw[H * IN];
    __shared__ float sb[H], sg[H], se[H], sp[H];
    __shared__ float sq[3];
    __shared__ float sstat[2 * H];

    const int b = threadIdx.x;
    const int t = blockIdx.x;

    for (int i = b; i < H * IN; i += 256) sw[i] = w[(size_t)t * H * IN + i];
    if (b < H) {
        sb[b] = bias[t*H + b]; sg[b] = gamma[t*H + b];
        se[b] = beta[t*H + b]; sp[b] = p[t*H + b];
    }
    if (b == 0) { sq[0] = q[t]; sq[1] = r[t]; sq[2] = s[t]; }
    if (b < 2*H) sstat[b] = 0.f;
    __syncthreads();

    const float* __restrict__ hrow = hin + (size_t)b * ((size_t)T * IN) + (size_t)t * IN;

    float z[H];
#pragma unroll
    for (int h = 0; h < H; ++h) {
        float a = sb[h];
#pragma unroll 8
        for (int i = 0; i < IN; ++i) a += sw[h*IN + i] * hrow[i];
        z[h] = tanhf(a);
    }

    // batch statistics: per-wave tree reduce, then LDS float atomics
#pragma unroll
    for (int h = 0; h < H; ++h) {
        float v = z[h], vs = z[h] * z[h];
        for (int off = 16; off > 0; off >>= 1) {
            v  += __shfl_xor(v,  off, 32);
            vs += __shfl_xor(vs, off, 32);
        }
        if ((b & 31) == 0) { atomicAdd(&sstat[h], v); atomicAdd(&sstat[H + h], vs); }
    }
    __syncthreads();

    float y[H];
#pragma unroll
    for (int h = 0; h < H; ++h) {
        float m   = sstat[h] * (1.f / 256.f);
        float var = sstat[H + h] * (1.f / 256.f) - m * m;
        y[h] = (z[h] - m) * rsqrtf(var + BN_EPS) * sg[h] + se[h];
        hout[(size_t)b * ((size_t)T * H) + (size_t)t * H + h] = y[h];
    }

    float a = sq[0];
#pragma unroll
    for (int h = 0; h < H; ++h) a += y[h] * sp[h];
    out[(size_t)b * OUTW + auxcol + t] = sq[1] * tanhf(a) + sq[2];
}

// ---------------------------------------------------------------------------
// Kernel 3: final 6->12 linear + tanh + BN + final aux head (column 2337)
// ---------------------------------------------------------------------------
__global__ __launch_bounds__(256) void final_kernel(
    const float* __restrict__ yroot,  // [256, 6]
    const float* __restrict__ fw, const float* __restrict__ fb,
    const float* __restrict__ fg, const float* __restrict__ fe,
    const float* __restrict__ pw, const float* __restrict__ pb,
    const float* __restrict__ rw, const float* __restrict__ rb,
    float* __restrict__ out)
{
    __shared__ float sstat[24];
    const int b = threadIdx.x;
    if (b < 24) sstat[b] = 0.f;

    float yr[6];
#pragma unroll
    for (int i = 0; i < 6; ++i) yr[i] = yroot[b*6 + i];

    float o[12];
#pragma unroll
    for (int j = 0; j < 12; ++j) {
        float a = fb[j];
#pragma unroll
        for (int i = 0; i < 6; ++i) a += yr[i] * fw[j*6 + i];
        o[j] = tanhf(a);
    }
    __syncthreads();

#pragma unroll
    for (int j = 0; j < 12; ++j) {
        float v = o[j], vs = o[j] * o[j];
        for (int off = 16; off > 0; off >>= 1) {
            v  += __shfl_xor(v,  off, 32);
            vs += __shfl_xor(vs, off, 32);
        }
        if ((b & 31) == 0) { atomicAdd(&sstat[j], v); atomicAdd(&sstat[12 + j], vs); }
    }
    __syncthreads();

    float a = pb[0];
#pragma unroll
    for (int j = 0; j < 12; ++j) {
        float m   = sstat[j] * (1.f / 256.f);
        float var = sstat[12 + j] * (1.f / 256.f) - m * m;
        float of  = (o[j] - m) * rsqrtf(var + BN_EPS) * fg[j] + fe[j];
        a += of * pw[j];
    }
    out[(size_t)b * OUTW + 2337] = tanhf(a) * rw[0] + rb[0];
}

// ---------------------------------------------------------------------------
// Launch
// ---------------------------------------------------------------------------
extern "C" void kernel_launch(void* const* d_in, const int* in_sizes, int n_in,
                              void* d_out, int out_size, void* d_ws, size_t ws_size,
                              hipStream_t stream) {
    (void)in_sizes; (void)n_in; (void)out_size; (void)ws_size;

    const float* x  = (const float*)d_in[0];
    const float* gW = (const float*)d_in[1];
    const float* gb = (const float*)d_in[2];
    // per-level params: base = 3 + 8*l : w,b,g,e,p,q,r,s
    const float* P[4][8];
    for (int l = 0; l < 4; ++l)
        for (int i = 0; i < 8; ++i)
            P[l][i] = (const float*)d_in[3 + 8*l + i];
    const float* fw = (const float*)d_in[35];
    const float* fb = (const float*)d_in[36];
    const float* fg = (const float*)d_in[37];
    const float* fe = (const float*)d_in[38];
    const float* pw = (const float*)d_in[39];
    const float* pb = (const float*)d_in[40];
    const float* rw = (const float*)d_in[41];
    const float* rb = (const float*)d_in[42];

    float* out = (float*)d_out;
    float* ws  = (float*)d_ws;
    float* h0 = ws;                                   // [256, 20480]
    float* y0 = h0 + (size_t)BATCH * NTOT;            // [256, 12288]
    float* y1 = y0 + (size_t)BATCH * 12288;           // [256, 1536]
    float* y2 = y1 + (size_t)BATCH * 1536;            // [256, 192]
    float* y3 = y2 + (size_t)BATCH * 192;             // [256, 6]
    uint32_t* xbf = (uint32_t*)(y3 + (size_t)BATCH * 6);  // [256*3008] bf16

    xcvt_kernel<<<(BATCH * NGENE) / (256 * 8), 256, 0, stream>>>(x, xbf);

    gene_gemm_kernel<<<NTOT / 128, 256, 0, stream>>>(xbf, gW, gb, h0);

    level_kernel<10><<<2048, 256, 0, stream>>>(h0,
        P[0][0], P[0][1], P[0][2], P[0][3], P[0][4], P[0][5], P[0][6], P[0][7],
        y0, out, 2048, 0);
    level_kernel<48><<<256, 256, 0, stream>>>(y0,
        P[1][0], P[1][1], P[1][2], P[1][3], P[1][4], P[1][5], P[1][6], P[1][7],
        y1, out, 256, 2048);
    level_kernel<48><<<32, 256, 0, stream>>>(y1,
        P[2][0], P[2][1], P[2][2], P[2][3], P[2][4], P[2][5], P[2][6], P[2][7],
        y2, out, 32, 2304);
    level_kernel<192><<<1, 256, 0, stream>>>(y2,
        P[3][0], P[3][1], P[3][2], P[3][3], P[3][4], P[3][5], P[3][6], P[3][7],
        y3, out, 1, 2336);

    final_kernel<<<1, 256, 0, stream>>>(y3, fw, fb, fg, fe, pw, pb, rw, rb, out);
}
